// MultiHeadSelfAttention_37701222924889
// MI455X (gfx1250) — matrix operand order repaired and checked by reading, not compiled, not run on hardware
//
#include <hip/hip_runtime.h>
#include <hip/hip_bf16.h>
#include <cstdint>

typedef __attribute__((ext_vector_type(16))) __bf16 bf16x16;
typedef __attribute__((ext_vector_type(8)))  float  floatx8;
typedef __attribute__((ext_vector_type(4)))  __bf16 bf16x4;
typedef __attribute__((ext_vector_type(4)))  unsigned int u32x4;
typedef __attribute__((ext_vector_type(8)))  int i32x8;
typedef __attribute__((ext_vector_type(4)))  int i32x4;

#if __has_builtin(__builtin_amdgcn_tensor_load_to_lds)
#define USE_TDM 1
#else
#define USE_TDM 0
#endif

union FragU { uint4 u[2]; bf16x16 v; };

__device__ __forceinline__ floatx8 wmma_bf16(bf16x16 a, bf16x16 b, floatx8 c) {
  return __builtin_amdgcn_wmma_f32_16x16x32_bf16(false, a, false, b, (short)0, c,
                                                 false, false);
}

// A-operand (16x32 bf16, row-major row of >=32 elements):
//   lanes 0-15: v0..3 = K0..7, v4..7 = K16..23 ; lanes 16-31: K8..15 / K24..31
__device__ __forceinline__ bf16x16 load_frag_a(const __bf16* row, int hf) {
  FragU f;
  f.u[0] = *(const uint4*)(row + hf * 8);
  f.u[1] = *(const uint4*)(row + 16 + hf * 8);
  return f.v;
}
// B-operand (32x16 bf16, N-major row of >=32 elements):
//   lanes 0-15: v0..3 = K0..7, v4..7 = K16..23 ; lanes 16-31: K8..15 / K24..31
__device__ __forceinline__ bf16x16 load_frag_b(const __bf16* row, int hf) {
  FragU f;
  f.u[0] = *(const uint4*)(row + hf * 8);
  f.u[1] = *(const uint4*)(row + 16 + hf * 8);
  return f.v;
}

// ---- DPP 16-lane row reductions (no LDS traffic, single VALU op per step) ---
template <int CTRL>
__device__ __forceinline__ float dpp_xor(float x) {
  return __builtin_bit_cast(
      float, __builtin_amdgcn_update_dpp(0, __builtin_bit_cast(int, x), CTRL,
                                         0xF, 0xF, true));
}
// 0xB1 = quad_perm(1,0,3,2) xor1 ; 0x4E = quad_perm(2,3,0,1) xor2
// 0x141 = row_half_mirror (xor4 once quads uniform) ; 0x140 = row_mirror (xor8)
__device__ __forceinline__ float row_max16(float x) {
  x = fmaxf(x, dpp_xor<0xB1>(x));
  x = fmaxf(x, dpp_xor<0x4E>(x));
  x = fmaxf(x, dpp_xor<0x141>(x));
  x = fmaxf(x, dpp_xor<0x140>(x));
  return x;
}
__device__ __forceinline__ float row_sum16(float x) {
  x += dpp_xor<0xB1>(x);
  x += dpp_xor<0x4E>(x);
  x += dpp_xor<0x141>(x);
  x += dpp_xor<0x140>(x);
  return x;
}

// ---------------------------------------------------------------------------
__global__ __launch_bounds__(256) void pack_bf16(const float* __restrict__ in,
                                                 __bf16* __restrict__ out, int n4) {
  int i = blockIdx.x * blockDim.x + threadIdx.x;
  if (i >= n4) return;
  float4 f = ((const float4*)in)[i];
  bf16x4 v = {(__bf16)f.x, (__bf16)f.y, (__bf16)f.z, (__bf16)f.w};
  *(bf16x4*)(out + (size_t)i * 4) = v;
}

// fp32 [R][C] -> bf16 [C][R] (N-major weights for WMMA B operands)
__global__ __launch_bounds__(256) void transpose_bf16(const float* __restrict__ in,
                                                      __bf16* __restrict__ out,
                                                      int R, int C) {
  __shared__ float tile[32][33];
  const int tx = threadIdx.x, ty = threadIdx.y;
  const int x = blockIdx.x * 32 + tx;
  const int y0 = blockIdx.y * 32 + ty;
#pragma unroll
  for (int i = 0; i < 4; ++i)
    tile[ty + i * 8][tx] = in[(size_t)(y0 + i * 8) * C + x];
  __syncthreads();
  const int ox = blockIdx.y * 32 + tx;
  const int oy0 = blockIdx.x * 32 + ty;
#pragma unroll
  for (int i = 0; i < 4; ++i)
    out[(size_t)(oy0 + i * 8) * R + ox] = (__bf16)tile[tx][ty + i * 8];
}

// ---------------------------------------------------------------------------
// Tiled bf16 GEMM: C[M=rows][N] = A[M][1024] * Bt[N][1024]^T + bias
// TDM stages the 128x32 tiles (pad_enable -> 80B LDS pitch), double-buffered.
// MODE 0: QKV projection -> scatter into Q (scaled 1/8), K, V^T  (bf16)
// MODE 1: output projection -> fp32 C
template <int MODE>
__global__ __launch_bounds__(256)
void gemm_bf16k(const __bf16* __restrict__ A, const __bf16* __restrict__ Bt,
                const float* __restrict__ bias, int NB, __bf16* __restrict__ Qo,
                __bf16* __restrict__ Ko, __bf16* __restrict__ Vto,
                float* __restrict__ Co) {
  constexpr int TILE = 128 * 40;  // elements per buffer (pitch 40)
  __shared__ alignas(16) __bf16 As[2 * TILE];
  __shared__ alignas(16) __bf16 Bs[2 * TILE];
  const int tid = threadIdx.x;
  const int lane = tid & 31;
  const int hf = lane >> 4, ln = lane & 15;
  const int wv = tid >> 5;
  const int wm = wv & 3, wn = wv >> 2;  // 4x2 wave grid, 32x64 per wave
  const int rowBase = blockIdx.y * 128;
  const int colBase = blockIdx.x * 128;

  floatx8 acc[2][4] = {};

#if USE_TDM
  // D# group1: data_size=2B, pad_enable, pad 16B every 64B (pitch 40 el)
  const int padw = (int)((1u << 16) | (1u << 20) | (3u << 22) | (3u << 25));
  const i32x8 g1A = {padw,
                     (int)(1024u << 16),  // tensor_dim0 = 1024 (K)
                     (int)(4096u << 16),  // tensor_dim1 = 4096 rows
                     (int)(32u << 16),    // tile_dim0   = 32
                     128,                 // tile_dim1   = 128
                     1024,                // tensor_dim0_stride
                     0, 0};
  const i32x8 g1B = {padw,
                     (int)(1024u << 16),          // tensor_dim0 = 1024 (K)
                     (int)(((unsigned)NB) << 16), // tensor_dim1 = N rows
                     (int)(32u << 16),            // tile_dim0   = 32
                     128,                         // tile_dim1   = 128
                     1024,                        // tensor_dim0_stride
                     0, 0};
  const i32x4 z4 = {0, 0, 0, 0};
  const i32x8 z8 = {0, 0, 0, 0, 0, 0, 0, 0};
  const unsigned asLds = (unsigned)(uintptr_t)(void*)As;
  const unsigned bsLds = (unsigned)(uintptr_t)(void*)Bs;
  const unsigned long long aBase =
      (unsigned long long)(uintptr_t)(A + (size_t)rowBase * 1024);
  const unsigned long long bBase =
      (unsigned long long)(uintptr_t)(Bt + (size_t)colBase * 1024);
  auto issueAB = [&](int it, int p) {
    const unsigned long long ka = aBase + (unsigned long long)it * 64ull;  // 32 el
    const unsigned long long kb = bBase + (unsigned long long)it * 64ull;
    u32x4 g0A = {1u, asLds + (unsigned)(p * TILE * 2), (unsigned)ka,
                 (unsigned)((ka >> 32) & 0x1FFFFFFu) | (2u << 30)};
    u32x4 g0B = {1u, bsLds + (unsigned)(p * TILE * 2), (unsigned)kb,
                 (unsigned)((kb >> 32) & 0x1FFFFFFu) | (2u << 30)};
    __builtin_amdgcn_tensor_load_to_lds(g0A, g1A, z4, z4, z8, 0);
    __builtin_amdgcn_tensor_load_to_lds(g0B, g1B, z4, z4, z8, 0);
  };
  const bool issuer = __builtin_amdgcn_readfirstlane(tid) < 32;  // scalar branch
  if (issuer) issueAB(0, 0);
#endif

  for (int it = 0; it < 32; ++it) {
    const int p = it & 1;
#if USE_TDM
    if (issuer) {
      if (it < 31) {
        issueAB(it + 1, p ^ 1);
        __builtin_amdgcn_s_wait_tensorcnt(2);  // tile `it` landed (in-order)
      } else {
        __builtin_amdgcn_s_wait_tensorcnt(0);
      }
    }
#else
#pragma unroll
    for (int q = 0; q < 2; ++q) {  // 128 rows x 32 cols bf16, 16B per thread
      const int r = q * 64 + (tid >> 2);
      const int c = (tid & 3) * 8;
      *(uint4*)(As + p * TILE + r * 40 + c) =
          *(const uint4*)(A + (size_t)(rowBase + r) * 1024 + it * 32 + c);
      *(uint4*)(Bs + p * TILE + r * 40 + c) =
          *(const uint4*)(Bt + (size_t)(colBase + r) * 1024 + it * 32 + c);
    }
#endif
    __syncthreads();  // tiles for `it` ready

    const __bf16* Asb = As + p * TILE;
    const __bf16* Bsb = Bs + p * TILE;
    bf16x16 a0 = load_frag_a(Asb + (wm * 32 + ln) * 40, hf);
    bf16x16 a1 = load_frag_a(Asb + (wm * 32 + 16 + ln) * 40, hf);
#pragma unroll
    for (int nt = 0; nt < 4; ++nt) {
      bf16x16 b = load_frag_b(Bsb + (wn * 64 + nt * 16 + ln) * 40, hf);
      acc[0][nt] = wmma_bf16(a0, b, acc[0][nt]);
      acc[1][nt] = wmma_bf16(a1, b, acc[1][nt]);
    }
    __syncthreads();  // buffer p free for DMA of tile it+2
  }

#pragma unroll
  for (int nt = 0; nt < 4; ++nt) {
    const int n = colBase + wn * 64 + nt * 16 + ln;
    const float bv = bias[n];
#pragma unroll
    for (int mt = 0; mt < 2; ++mt) {
#pragma unroll
      for (int r = 0; r < 8; ++r) {
        const int row = rowBase + wm * 32 + mt * 16 + r + 8 * hf;
        const float val = acc[mt][nt][r] + bv;
        if constexpr (MODE == 0) {
          const int s = n >> 10, rem = n & 1023;
          const int h = rem >> 6, d = rem & 63;
          const int b = row >> 11, l = row & 2047;
          const size_t bh = (size_t)(b * 16 + h);
          if (s == 0)
            Qo[(bh * 2048 + l) * 64 + d] = (__bf16)(val * 0.125f);  // 1/sqrt(64)
          else if (s == 1)
            Ko[(bh * 2048 + l) * 64 + d] = (__bf16)val;
          else
            Vto[(bh * 64 + d) * 2048 + l] = (__bf16)val;  // V transposed
        } else {
          Co[(size_t)row * 1024 + n] = val;
        }
      }
    }
  }
}

// ---------------------------------------------------------------------------
// Flash attention: block = 8 waves, each wave owns 16 queries of one (b,h).
// K/V chunks staged by the TDM into double-buffered LDS (DMA of chunk kc+1
// overlaps WMMA+softmax of chunk kc); S and O live in WMMA accumulators.
__global__ __launch_bounds__(256)
void attn_kernel(const __bf16* __restrict__ Q, const __bf16* __restrict__ K,
                 const __bf16* __restrict__ Vt, __bf16* __restrict__ Ob) {
  constexpr int KT = 32 * 72;  // K tile elements (pitch 72)
  constexpr int VT = 64 * 40;  // V tile elements (pitch 40)
  __shared__ alignas(16) __bf16 Ks[2 * KT];
  __shared__ alignas(16) __bf16 Vs[2 * VT];
  __shared__ alignas(16) __bf16 Ps[8 * 16 * 40];  // per-wave P transpose tile
  const int tid = threadIdx.x;
  const int lane = tid & 31, wv = tid >> 5;
  const int hf = lane >> 4, ln = lane & 15;
  const int bh = blockIdx.y;  // b*16 + h
  const int qBase = blockIdx.x * 128 + wv * 16;

  // Q A-fragments (K-dim = d_head, two 32-wide steps); Q pre-scaled by 1/8
  const __bf16* Qp = Q + ((size_t)bh * 2048 + qBase + ln) * 64;
  FragU aq0, aq1;
  aq0.u[0] = *(const uint4*)(Qp + hf * 8);
  aq0.u[1] = *(const uint4*)(Qp + 16 + hf * 8);
  aq1.u[0] = *(const uint4*)(Qp + 32 + hf * 8);
  aq1.u[1] = *(const uint4*)(Qp + 48 + hf * 8);

  floatx8 o0 = {}, o1 = {}, o2 = {}, o3 = {};
  float rmax[8], rsum[8];
#pragma unroll
  for (int r = 0; r < 8; ++r) { rmax[r] = -__builtin_inff(); rsum[r] = 0.0f; }

  __bf16* Pw = Ps + wv * (16 * 40);

#if USE_TDM
  // K tile: tensor [2048][64], tile 32x64, pad 16B every 128B (pitch 72 el)
  const i32x8 g1K = {(int)((1u << 16) | (1u << 20) | (4u << 22) | (3u << 25)),
                     (int)(64u << 16),    // tensor_dim0 = 64
                     (int)(2048u << 16),  // tensor_dim1 = 2048
                     (int)(64u << 16),    // tile_dim0   = 64
                     32,                  // tile_dim1   = 32
                     64,                  // tensor_dim0_stride
                     0, 0};
  // V tile: tensor [64][2048], tile 64x32, pad 16B every 64B (pitch 40 el)
  const i32x8 g1V = {(int)((1u << 16) | (1u << 20) | (3u << 22) | (3u << 25)),
                     (int)(2048u << 16),  // tensor_dim0 = 2048
                     (int)(64u << 16),    // tensor_dim1 = 64
                     (int)(32u << 16),    // tile_dim0   = 32
                     64,                  // tile_dim1   = 64
                     2048,                // tensor_dim0_stride
                     0, 0};
  const i32x4 z4 = {0, 0, 0, 0};
  const i32x8 z8 = {0, 0, 0, 0, 0, 0, 0, 0};
  const unsigned ksLds = (unsigned)(uintptr_t)(void*)Ks;
  const unsigned vsLds = (unsigned)(uintptr_t)(void*)Vs;
  const unsigned long long kBase =
      (unsigned long long)(uintptr_t)(K + (size_t)bh * 2048 * 64);
  const unsigned long long vBase =
      (unsigned long long)(uintptr_t)(Vt + (size_t)bh * 64 * 2048);
  auto issueKV = [&](int kcc, int p) {
    const unsigned long long ka = kBase + (unsigned long long)kcc * (32ull * 128ull);
    const unsigned long long va = vBase + (unsigned long long)kcc * 64ull;
    u32x4 g0K = {1u, ksLds + (unsigned)(p * KT * 2), (unsigned)ka,
                 (unsigned)((ka >> 32) & 0x1FFFFFFu) | (2u << 30)};
    u32x4 g0V = {1u, vsLds + (unsigned)(p * VT * 2), (unsigned)va,
                 (unsigned)((va >> 32) & 0x1FFFFFFu) | (2u << 30)};
    __builtin_amdgcn_tensor_load_to_lds(g0K, g1K, z4, z4, z8, 0);
    __builtin_amdgcn_tensor_load_to_lds(g0V, g1V, z4, z4, z8, 0);
  };
  const bool issuer = __builtin_amdgcn_readfirstlane(tid) < 32;  // scalar branch
  if (issuer) issueKV(0, 0);
#endif

  for (int kc = 0; kc < 64; ++kc) {
    const int p = kc & 1;
#if USE_TDM
    if (issuer) {
      if (kc < 63) {
        issueKV(kc + 1, p ^ 1);
        __builtin_amdgcn_s_wait_tensorcnt(2);  // chunk kc landed (in-order)
      } else {
        __builtin_amdgcn_s_wait_tensorcnt(0);
      }
    }
#else
    {
      const int r = tid >> 3, c = (tid & 7) * 8;
      *(uint4*)(Ks + p * KT + r * 72 + c) =
          *(const uint4*)(K + ((size_t)bh * 2048 + kc * 32 + r) * 64 + c);
      const int r2 = tid >> 2, c2 = (tid & 3) * 8;
      *(uint4*)(Vs + p * VT + r2 * 40 + c2) =
          *(const uint4*)(Vt + ((size_t)bh * 64 + r2) * 2048 + kc * 32 + c2);
    }
#endif
    __syncthreads();  // chunk kc tiles ready

    const __bf16* Ksb = Ks + p * KT;
    const __bf16* Vsb = Vs + p * VT;

    // S(16x32) = Q(16x64) @ K_chunk^T : B-operand rows are K rows (N-major)
    floatx8 s0 = {}, s1 = {};
    s0 = wmma_bf16(aq0.v, load_frag_b(Ksb + ln * 72, hf), s0);
    s0 = wmma_bf16(aq1.v, load_frag_b(Ksb + ln * 72 + 32, hf), s0);
    s1 = wmma_bf16(aq0.v, load_frag_b(Ksb + (16 + ln) * 72, hf), s1);
    s1 = wmma_bf16(aq1.v, load_frag_b(Ksb + (16 + ln) * 72 + 32, hf), s1);

    // online softmax: row m = r + 8*hf lives across the 16 lanes of one half
#pragma unroll
    for (int r = 0; r < 8; ++r) {
      const float cmax = row_max16(fmaxf(s0[r], s1[r]));
      const float mnew = fmaxf(rmax[r], cmax);
      const float corr = __expf(rmax[r] - mnew);
      rmax[r] = mnew;
      const float p0 = __expf(s0[r] - mnew);
      const float p1 = __expf(s1[r] - mnew);
      const float csum = row_sum16(p0 + p1);
      rsum[r] = rsum[r] * corr + csum;
      o0[r] *= corr; o1[r] *= corr; o2[r] *= corr; o3[r] *= corr;
      const int m = r + 8 * hf;
      Pw[m * 40 + ln]      = (__bf16)p0;  // C-layout -> LDS -> A-layout
      Pw[m * 40 + 16 + ln] = (__bf16)p1;
    }

    // P A-fragment (16 queries x 32 keys); same-wave DS ops stay in order
    FragU ap;
    const __bf16* pp = Pw + ln * 40;
    ap.u[0] = *(const uint4*)(pp + hf * 8);
    ap.u[1] = *(const uint4*)(pp + 16 + hf * 8);

    // O(16x64) += P(16x32) @ V_chunk(32x64); B rows are V^T rows (N-major)
    o0 = wmma_bf16(ap.v, load_frag_b(Vsb + ln * 40, hf), o0);
    o1 = wmma_bf16(ap.v, load_frag_b(Vsb + (16 + ln) * 40, hf), o1);
    o2 = wmma_bf16(ap.v, load_frag_b(Vsb + (32 + ln) * 40, hf), o2);
    o3 = wmma_bf16(ap.v, load_frag_b(Vsb + (48 + ln) * 40, hf), o3);

    __syncthreads();  // buffer p free for DMA of chunk kc+2
  }

  const int b = bh >> 4, h = bh & 15;
#pragma unroll
  for (int r = 0; r < 8; ++r) {
    const float inv = 1.0f / rsum[r];
    const int row = qBase + r + 8 * hf;
    const size_t base = ((size_t)b * 2048 + row) * 1024 + h * 64 + ln;
    Ob[base]      = (__bf16)(o0[r] * inv);
    Ob[base + 16] = (__bf16)(o1[r] * inv);
    Ob[base + 32] = (__bf16)(o2[r] * inv);
    Ob[base + 48] = (__bf16)(o3[r] * inv);
  }
}

// ---------------------------------------------------------------------------
extern "C" void kernel_launch(void* const* d_in, const int* in_sizes, int n_in,
                              void* d_out, int out_size, void* d_ws, size_t ws_size,
                              hipStream_t stream) {
  (void)in_sizes; (void)n_in; (void)out_size; (void)ws_size;
  const float* x    = (const float*)d_in[0];
  const float* Wqkv = (const float*)d_in[1];
  const float* bqkv = (const float*)d_in[2];
  const float* Wout = (const float*)d_in[3];
  const float* bout = (const float*)d_in[4];
  float* out = (float*)d_out;

  char* ws = (char*)d_ws;                       // 48 MiB total
  __bf16* Xb    = (__bf16*)(ws);                // [4096][1024]      8 MiB
  __bf16* WqkvT = (__bf16*)(ws + (8u  << 20));  // [3072][1024]      6 MiB
  __bf16* WoutT = (__bf16*)(ws + (14u << 20));  // [1024][1024]      2 MiB
  __bf16* Qb    = (__bf16*)(ws + (16u << 20));  // [2][16][2048][64] 8 MiB (pre-scaled)
  __bf16* Kb    = (__bf16*)(ws + (24u << 20));  // [2][16][2048][64] 8 MiB
  __bf16* Vtb   = (__bf16*)(ws + (32u << 20));  // [2][16][64][2048] 8 MiB
  __bf16* Ob    = (__bf16*)(ws + (40u << 20));  // [4096][1024]      8 MiB

  pack_bf16<<<4096, 256, 0, stream>>>(x, Xb, 4096 * 1024 / 4);
  transpose_bf16<<<dim3(96, 32), dim3(32, 8), 0, stream>>>(Wqkv, WqkvT, 1024, 3072);
  transpose_bf16<<<dim3(32, 32), dim3(32, 8), 0, stream>>>(Wout, WoutT, 1024, 1024);
  gemm_bf16k<0><<<dim3(24, 32), 256, 0, stream>>>(Xb, WqkvT, bqkv, 3072, Qb, Kb,
                                                  Vtb, nullptr);
  attn_kernel<<<dim3(16, 32), 256, 0, stream>>>(Qb, Kb, Vtb, Ob);
  gemm_bf16k<1><<<dim3(8, 32), 256, 0, stream>>>(Ob, WoutT, bout, 1024, nullptr,
                                                 nullptr, nullptr, out);
}
